// SwinBlock_13726715478131
// MI455X (gfx1250) — matrix operand order repaired and checked by reading, not compiled, not run on hardware
//
#include <hip/hip_runtime.h>
#include <hip/hip_bf16.h>

// ---------------------------------------------------------------------------
// Swin block, fully fused, bf16 WMMA (gfx1250 / CDNA5, wave32).
//   grid = 8192 windows, block = 256 threads (8 wave32), dynamic LDS ~217 KB.
//   V is stored transposed per head so ALL wmma operands load as contiguous
//   16B LDS chunks (no scalar gathers).
// ---------------------------------------------------------------------------

typedef __bf16 bf16;
typedef __attribute__((ext_vector_type(16))) __bf16 v16bf;
typedef __attribute__((ext_vector_type(8)))  __bf16 v8bf;
typedef __attribute__((ext_vector_type(8)))  float  v8f;

#define RS   264   // resid/actA row stride (bf16): 528B, %16==0, bank-shift 4
#define QS   776   // qkv row stride (bf16): 1552B, %16==0, bank-shift 4
#define HS   72    // attS slab / hchunk / vT row stride: 144B, %16==0
#define FS   260   // f32 staging row stride

__device__ __forceinline__ v8f wmma_bf16(v16bf a, v16bf b, v8f c) {
  return __builtin_amdgcn_wmma_f32_16x16x32_bf16(false, a, false, b, (short)0, c,
                                                 false, false);
}

// A fragment (16x32 bf16) from LDS, row-major [m][k], stride in elements.
// lanes 0-15: m = m0+lane, K = k0+0..15; lanes 16-31: K = k0+16..31
__device__ __forceinline__ v16bf ldsA(const bf16* p, int stride, int m0, int k0) {
  int lane = threadIdx.x & 31, hf = lane >> 4, l = lane & 15;
  const bf16* row = p + (m0 + l) * stride + k0 + hf * 16;
  v8bf c0 = *(const v8bf*)(row);
  v8bf c1 = *(const v8bf*)(row + 8);
  v16bf a;
#pragma unroll
  for (int i = 0; i < 8; ++i) { a[i] = c0[i]; a[i + 8] = c1[i]; }
  return a;
}

// B fragment (32x16) when memory holds B^T rows: [n][k] row-major
// (K for Q@K^T, and transposed V for att@V).
// lanes 0-15: n = n0+lane, k = k0+0..15 ; lanes 16-31: same n, k = k0+16..31
__device__ __forceinline__ v16bf ldsBrows(const bf16* p, int stride, int n0, int k0) {
  int lane = threadIdx.x & 31, hf = lane >> 4, l = lane & 15;
  const bf16* row = p + (n0 + l) * stride + k0 + hf * 16;
  v8bf c0 = *(const v8bf*)(row);
  v8bf c1 = *(const v8bf*)(row + 8);
  v16bf b;
#pragma unroll
  for (int i = 0; i < 8; ++i) { b[i] = c0[i]; b[i + 8] = c1[i]; }
  return b;
}

// Pre-packed global B fragment: 512 bf16 per 16x32-k tile, lane-contiguous 32B.
__device__ __forceinline__ v16bf gB(const bf16* frags, int tile) {
  return ((const v16bf*)frags)[tile * 32 + (threadIdx.x & 31)];
}

__device__ __forceinline__ float wsum32(float v) {
#pragma unroll
  for (int m = 16; m >= 1; m >>= 1) v += __shfl_xor(v, m, 32);
  return v;
}
__device__ __forceinline__ float hmax16(float v) {
#pragma unroll
  for (int m = 8; m >= 1; m >>= 1) v = fmaxf(v, __shfl_xor(v, m, 32));
  return v;
}
__device__ __forceinline__ float hsum16(float v) {
#pragma unroll
  for (int m = 8; m >= 1; m >>= 1) v += __shfl_xor(v, m, 32);
  return v;
}

// ---------------------------------------------------------------------------
// Weight pre-pack: f32 W[N][K] -> bf16 fragments of B = W^T (K x N).
// tile order [nt][kt]; within tile: lane*16 + e, n = nt*16 + lane%16,
// k = kt*32 + (lane/16)*16 + e.
// ---------------------------------------------------------------------------
__global__ __launch_bounds__(256) void swin_prepack(const float* __restrict__ W,
                                                    bf16* __restrict__ dst,
                                                    int K, int N) {
  int idx = blockIdx.x * 256 + threadIdx.x;
  if (idx >= K * N) return;
  int e = idx & 15;
  int l = (idx >> 4) & 31;
  int tile = idx >> 9;
  int KT = K >> 5;
  int nt = tile / KT, kt = tile - nt * KT;
  int n = nt * 16 + (l & 15);
  int k = kt * 32 + ((l >> 4) << 4) + e;
  dst[idx] = (bf16)W[n * K + k];
}

// ---------------------------------------------------------------------------
// Main fused Swin-block kernel. One workgroup per 7x7 window.
// ---------------------------------------------------------------------------
__global__ __launch_bounds__(256) void swin_block(
    const float* __restrict__ x, const float* __restrict__ n1w,
    const float* __restrict__ n1b, const float* __restrict__ ipb,
    const float* __restrict__ opb, const float* __restrict__ n2w,
    const float* __restrict__ n2b, const float* __restrict__ m1b,
    const float* __restrict__ m2b, const bf16* __restrict__ wQKV,
    const bf16* __restrict__ wOut, const bf16* __restrict__ wM1,
    const bf16* __restrict__ wM2, float* __restrict__ out) {
  extern __shared__ char smem[];
  bf16* resid = (bf16*)smem;          // [64][RS]  residual (LN1 output, then updated)
  bf16* actA  = resid + 64 * RS;      // [64][RS]  generic A-operand buffer
  bf16* qkvS  = actA + 64 * RS;       // [64][QS]  q|k (cols 0..511); overlays below
  bf16* attS  = qkvS + 64 * QS;       // [8][16][HS] per-wave softmax slabs
  bf16* vT    = attS + 8 * 16 * HS;   // [8][32][HS] V transposed per head
  float* rawS = (float*)qkvS;         // [49][FS] raw input staging
  float* outS = (float*)qkvS;         // [49][FS] final output staging
  bf16* hch   = qkvS;                 // [64][HS] MLP hidden chunk

  const int tid = threadIdx.x, lane = tid & 31, wv = tid >> 5;
  const int hf = lane >> 4, l15 = lane & 15;

  const int bi = blockIdx.x;
  const int b = bi >> 10, rem = bi & 1023, wy = rem >> 5, wx = rem & 31;
  const long HW = 224L * 224L;
  const long base = (long)b * 256 * HW + (long)wy * 7 * 224 + (long)wx * 7;
  const float* xb = x + base;
  float* ob = out + base;

  const v8f zf = {0.f, 0.f, 0.f, 0.f, 0.f, 0.f, 0.f, 0.f};

  // ---------------- Phase 0: load window (coalesced-ish) + LayerNorm1 -------
  for (int i = tid; i < 49 * 256; i += 256) {
    int c = i / 49, p = i - c * 49;
    int y = p / 7, xq = p - y * 7;
    rawS[p * FS + c] = xb[(long)c * HW + y * 224 + xq];
  }
  __syncthreads();
  for (int t = wv; t < 49; t += 8) {  // wave per token
    float s = 0.f, s2 = 0.f;
    for (int c = lane; c < 256; c += 32) {
      float v = rawS[t * FS + c];
      s += v; s2 += v * v;
    }
    s = wsum32(s); s2 = wsum32(s2);
    float mean = s * (1.f / 256.f);
    float inv = rsqrtf(s2 * (1.f / 256.f) - mean * mean + 1e-5f);
    for (int c = lane; c < 256; c += 32) {
      float v = (rawS[t * FS + c] - mean) * inv * n1w[c] + n1b[c];
      resid[t * RS + c] = (bf16)v;
    }
  }
  for (int i = tid; i < 15 * RS; i += 256) resid[49 * RS + i] = (bf16)0.f;
  __syncthreads();

  // ---------------- Phase 1: QKV = resid @ Wqkv^T + b  (64x256 @ 256x768) ---
  // Q,K (cols 0..511) -> qkvS ; V (cols 512..767) -> vT[head][ch][token].
  for (int tile = wv; tile < 192; tile += 8) {
    int mt = tile & 3, nt = tile >> 2;
    v8f acc = zf;
#pragma unroll
    for (int kt = 0; kt < 8; ++kt) {
      if (kt < 7) __builtin_prefetch((const void*)(((const v16bf*)wQKV) +
                                     (nt * 8 + kt + 1) * 32 + lane), 0, 3);
      v16bf a = ldsA(resid, RS, mt * 16, kt * 32);
      acc = wmma_bf16(a, gB(wQKV, nt * 8 + kt), acc);
    }
    int col = nt * 16 + l15, m0 = mt * 16 + hf * 8;
    float bias = ipb[col];
    if (nt < 32) {  // Q or K: token-major
#pragma unroll
      for (int r = 0; r < 8; ++r) qkvS[(m0 + r) * QS + col] = (bf16)(acc[r] + bias);
    } else {        // V: transposed, per-head [ch][token]
      int cg = col - 512;
      bf16* dv = vT + (cg >> 5) * (32 * HS) + (cg & 31) * HS;
#pragma unroll
      for (int r = 0; r < 8; ++r) dv[m0 + r] = (bf16)(acc[r] + bias);
    }
  }
  __syncthreads();

  // ---------------- Phase 2: attention, head = wave ------------------------
  {
    const int qc = wv * 32, kc = 256 + wv * 32;
    bf16* slab = attS + wv * (16 * HS);
    const bf16* vh = vT + wv * (32 * HS);
    const float SC = 0.17677669529663687f;  // 1/sqrt(32)
    for (int mt = 0; mt < 4; ++mt) {
      v16bf aq = ldsA(qkvS + qc, QS, mt * 16, 0);
      v8f s0 = wmma_bf16(aq, ldsBrows(qkvS + kc, QS, 0, 0), zf);
      v8f s1 = wmma_bf16(aq, ldsBrows(qkvS + kc, QS, 16, 0), zf);
      v8f s2 = wmma_bf16(aq, ldsBrows(qkvS + kc, QS, 32, 0), zf);
      v8f s3 = wmma_bf16(aq, ldsBrows(qkvS + kc, QS, 48, 0), zf);
#pragma unroll
      for (int r = 0; r < 8; ++r) {
        float v0 = s0[r] * SC, v1 = s1[r] * SC, v2 = s2[r] * SC;
        float v3 = (l15 == 0) ? s3[r] * SC : -1e30f;  // keys 49..63 masked
        float mx = hmax16(fmaxf(fmaxf(v0, v1), fmaxf(v2, v3)));
        float e0 = __expf(v0 - mx), e1 = __expf(v1 - mx);
        float e2 = __expf(v2 - mx), e3 = __expf(v3 - mx);
        float is = 1.f / hsum16(e0 + e1 + e2 + e3);
        int rr = (r + 8 * hf) * HS + l15;
        slab[rr]      = (bf16)(e0 * is);
        slab[rr + 16] = (bf16)(e1 * is);
        slab[rr + 32] = (bf16)(e2 * is);
        slab[rr + 48] = (bf16)(e3 * is);
      }
      __syncthreads();
      v8f oa0 = zf, oa1 = zf;
#pragma unroll
      for (int kk = 0; kk < 2; ++kk) {
        v16bf a = ldsA(slab, HS, 0, kk * 32);
        oa0 = wmma_bf16(a, ldsBrows(vh, HS, 0, kk * 32), oa0);
        oa1 = wmma_bf16(a, ldsBrows(vh, HS, 16, kk * 32), oa1);
      }
      int m0 = mt * 16 + 8 * hf;
#pragma unroll
      for (int r = 0; r < 8; ++r) {
        actA[(m0 + r) * RS + wv * 32 + l15]      = (bf16)oa0[r];
        actA[(m0 + r) * RS + wv * 32 + 16 + l15] = (bf16)oa1[r];
      }
      __syncthreads();
    }
  }

  // ---------------- Phase 3: out_proj + residual, then LN2 -----------------
  for (int tile = wv; tile < 64; tile += 8) {
    int mt = tile & 3, nt = tile >> 2;
    v8f acc = zf;
#pragma unroll
    for (int kt = 0; kt < 8; ++kt)
      acc = wmma_bf16(ldsA(actA, RS, mt * 16, kt * 32), gB(wOut, nt * 8 + kt), acc);
    int col = nt * 16 + l15, m0 = mt * 16 + 8 * hf;
    float bias = opb[col];
#pragma unroll
    for (int r = 0; r < 8; ++r) {
      int m = m0 + r;
      float v = acc[r] + bias + (float)resid[m * RS + col];
      resid[m * RS + col] = (bf16)v;
    }
  }
  __syncthreads();
  for (int t = wv; t < 49; t += 8) {  // LN2: resid -> actA
    float s = 0.f, s2 = 0.f;
    for (int c = lane; c < 256; c += 32) {
      float v = (float)resid[t * RS + c];
      s += v; s2 += v * v;
    }
    s = wsum32(s); s2 = wsum32(s2);
    float mean = s * (1.f / 256.f);
    float inv = rsqrtf(s2 * (1.f / 256.f) - mean * mean + 1e-5f);
    for (int c = lane; c < 256; c += 32) {
      float v = ((float)resid[t * RS + c] - mean) * inv * n2w[c] + n2b[c];
      actA[t * RS + c] = (bf16)v;
    }
  }
  __syncthreads();

  // ---------------- Phase 4: MLP fused over 16 chunks of 64 hidden cols ----
  v8f oacc[4][2];
#pragma unroll
  for (int mt = 0; mt < 4; ++mt) { oacc[mt][0] = zf; oacc[mt][1] = zf; }

  for (int chn = 0; chn < 16; ++chn) {
#pragma unroll
    for (int j = 0; j < 2; ++j) {  // this wave's 2 tiles of hidden chunk
      int id = wv * 2 + j, mt = id >> 2, ntc = id & 3;
      v8f acc = zf;
#pragma unroll
      for (int kt = 0; kt < 8; ++kt)
        acc = wmma_bf16(ldsA(actA, RS, mt * 16, kt * 32),
                        gB(wM1, (chn * 4 + ntc) * 8 + kt), acc);
      float bias = m1b[chn * 64 + ntc * 16 + l15];
      int m0 = mt * 16 + 8 * hf;
#pragma unroll
      for (int r = 0; r < 8; ++r) {
        float xv = acc[r] + bias;
        float g = 0.5f * xv * (1.f + erff(xv * 0.70710678118654752f));  // exact GELU
        hch[(m0 + r) * HS + ntc * 16 + l15] = (bf16)g;
      }
    }
    __syncthreads();
#pragma unroll
    for (int mt = 0; mt < 4; ++mt)
#pragma unroll
      for (int kk = 0; kk < 2; ++kk) {
        v16bf a = ldsA(hch, HS, mt * 16, kk * 32);
        oacc[mt][0] = wmma_bf16(a, gB(wM2, (wv * 2 + 0) * 32 + chn * 2 + kk), oacc[mt][0]);
        oacc[mt][1] = wmma_bf16(a, gB(wM2, (wv * 2 + 1) * 32 + chn * 2 + kk), oacc[mt][1]);
      }
    __syncthreads();
  }

  // Final: + mlp_b2 + residual -> f32 staging (overlays hch region, post-barrier)
#pragma unroll
  for (int mt = 0; mt < 4; ++mt)
#pragma unroll
    for (int ntl = 0; ntl < 2; ++ntl) {
      int col = wv * 32 + ntl * 16 + l15, m0 = mt * 16 + 8 * hf;
      float bias = m2b[col];
#pragma unroll
      for (int r = 0; r < 8; ++r) {
        int m = m0 + r;
        if (m < 49)
          outS[m * FS + col] = oacc[mt][ntl][r] + bias + (float)resid[m * RS + col];
      }
    }
  __syncthreads();
  for (int i = tid; i < 49 * 256; i += 256) {
    int c = i / 49, p = i - c * 49;
    int y = p / 7, xq = p - y * 7;
    ob[(long)c * HW + y * 224 + xq] = outS[p * FS + c];
  }
}

// ---------------------------------------------------------------------------
extern "C" void kernel_launch(void* const* d_in, const int* in_sizes, int n_in,
                              void* d_out, int out_size, void* d_ws, size_t ws_size,
                              hipStream_t stream) {
  (void)in_sizes; (void)n_in; (void)out_size; (void)ws_size;
  const float* x   = (const float*)d_in[0];
  const float* n1w = (const float*)d_in[1];
  const float* n1b = (const float*)d_in[2];
  const float* ipw = (const float*)d_in[3];
  const float* ipb = (const float*)d_in[4];
  const float* opw = (const float*)d_in[5];
  const float* opb = (const float*)d_in[6];
  const float* n2w = (const float*)d_in[7];
  const float* n2b = (const float*)d_in[8];
  const float* m1w = (const float*)d_in[9];
  const float* m1b = (const float*)d_in[10];
  const float* m2w = (const float*)d_in[11];
  const float* m2b = (const float*)d_in[12];
  float* outp = (float*)d_out;

  bf16* wQKV = (bf16*)d_ws;                 // 768x256  = 196608 elems
  bf16* wOut = wQKV + 196608;               // 256x256  =  65536
  bf16* wM1  = wOut + 65536;                // 1024x256 = 262144
  bf16* wM2  = wM1 + 262144;                // 256x1024 = 262144  (total ~1.5 MB)

  swin_prepack<<<768, 256, 0, stream>>>(ipw, wQKV, 256, 768);
  swin_prepack<<<256, 256, 0, stream>>>(opw, wOut, 256, 256);
  swin_prepack<<<1024, 256, 0, stream>>>(m1w, wM1, 256, 1024);
  swin_prepack<<<1024, 256, 0, stream>>>(m2w, wM2, 1024, 256);

  // dynamic LDS: (2*64*RS + 64*QS + 8*16*HS + 8*32*HS) bf16 = 222,208 bytes
  const size_t smem = (size_t)(2 * 64 * RS + 64 * QS + 8 * 16 * HS + 8 * 32 * HS) * 2;
  swin_block<<<8192, 256, smem, stream>>>(x, n1w, n1b, ipb, opb, n2w, n2b, m1b,
                                          m2b, wQKV, wOut, wM1, wM2, outp);
}
